// Decoder_34514357191048
// MI455X (gfx1250) — matrix operand order repaired and checked by reading, not compiled, not run on hardware
//
#include <hip/hip_runtime.h>
#include <hip/hip_bf16.h>
#include <math.h>
#include <stdint.h>

typedef __bf16 bf16_t;
typedef __attribute__((ext_vector_type(16))) __bf16 v16bf;
typedef __attribute__((ext_vector_type(8)))  float  v8f;
typedef uint32_t u32;

static constexpr int kB   = 1024;          // batch
static constexpr int kT   = 127;           // timesteps (T-1)
static constexpr int kM   = 256;           // encoder hidden
static constexpr int kP   = 256;           // decoder hidden
static constexpr int kNT  = kB * kT;       // 130048 encoder rows
static constexpr int kKW1 = 2 * kP + kM;   // 768 = attn_W1 inner dim

// ---------- bf16 helpers (manual RNE conversion, no arithmetic on __bf16) ----------
__device__ __forceinline__ bf16_t f2bf(float f) {
    union { float f; u32 u; } c; c.f = f;
    u32 r = c.u + 0x7FFFu + ((c.u >> 16) & 1u);
    union { unsigned short s; bf16_t h; } o; o.s = (unsigned short)(r >> 16);
    return o.h;
}
__device__ __forceinline__ float bf2f(bf16_t h) {
    union { unsigned short s; bf16_t h; } i; i.h = h;
    union { u32 u; float f; } c; c.u = ((u32)i.s) << 16;
    return c.f;
}
__device__ __forceinline__ float sigf(float x) { return 1.0f / (1.0f + __expf(-x)); }

__device__ __forceinline__ v8f wmma_bf16(v16bf a, v16bf b, v8f c) {
    // (neg_a, A, neg_b, B, c_mod, C, reuse_a, reuse_b)
    return __builtin_amdgcn_wmma_f32_16x16x32_bf16(false, a, false, b, (short)0, c, false, false);
}

// A tile 16x32 bf16 from row-major memory, row stride lda (elements).
// Layout: lanes 0-15 row m, K=0..15; lanes 16-31: K=16..31 (matches packed B).
__device__ __forceinline__ v16bf load_A_tile(const bf16_t* A, int m0, int k0, int lda, int lane) {
    int m = lane & 15;
    int kbase = k0 + ((lane & 16) ? 16 : 0);
    const bf16_t* row = A + (size_t)(m0 + m) * lda + kbase;
    union { uint4 u[2]; v16bf v; } cvt;
    cvt.u[0] = *(const uint4*)(row);
    cvt.u[1] = *(const uint4*)(row + 8);
    return cvt.v;
}

// B tile 32x16 bf16 from pre-packed buffer: dword (kt,nt,lane,j) holds {B[k+1],B[k]},
// k = kt*32 + (lane<16?0:16) + 2j, n = nt*16 + lane%16.
__device__ __forceinline__ v16bf load_B_tile(const u32* packed, int kt, int nt, int numNT, int lane) {
    const u32* p = packed + (((size_t)(kt * numNT + nt) * 32) + lane) * 8;
    union { uint4 u[2]; v16bf v; } cvt;
    cvt.u[0] = *(const uint4*)(p);
    cvt.u[1] = *(const uint4*)(p + 4);
    return cvt.v;
}

// ---------- setup kernels ----------
__global__ __launch_bounds__(256) void k_convert_bf16(const float* __restrict__ src,
                                                      bf16_t* __restrict__ dst, int n) {
    int i = blockIdx.x * 256 + threadIdx.x;
    if (i < n) dst[i] = f2bf(src[i]);
}

// Pack weight W (row-major [N][K] f32, element B[k][n] = W[n][k]) into WMMA-B layout.
__global__ __launch_bounds__(256) void k_pack_B(const float* __restrict__ W,
                                                u32* __restrict__ packed, int K, int N) {
    int idx = blockIdx.x * 256 + threadIdx.x;
    int total = (K >> 1) * N;  // dwords
    if (idx >= total) return;
    int j    = idx & 7;
    int lane = (idx >> 3) & 31;
    int tile = idx >> 8;
    int numNT = N >> 4;
    int nt = tile % numNT;
    int kt = tile / numNT;
    int n  = nt * 16 + (lane & 15);
    int k  = kt * 32 + ((lane & 16) ? 16 : 0) + 2 * j;
    union { bf16_t h[2]; u32 u; } c;
    c.h[0] = f2bf(W[(size_t)n * K + k]);
    c.h[1] = f2bf(W[(size_t)n * K + k + 1]);
    packed[idx] = c.u;
}

// U[n][h] = bf16( sum_k Xe[n][k]*W1c[h][k] + b1[h] )   (step-invariant precompute)
__global__ __launch_bounds__(256) void k_gemm_U(const bf16_t* __restrict__ Xe,
                                                const u32* __restrict__ pW1,
                                                const float* __restrict__ b1,
                                                bf16_t* __restrict__ U) {
    int gw = (blockIdx.x * 256 + threadIdx.x) >> 5;
    int lane = threadIdx.x & 31;
    const int numHT = kM / 16;
    int nt = gw / numHT;
    int ht = gw - nt * numHT;
    if (nt * 16 >= kNT) return;
    v8f acc = {};
    #pragma unroll
    for (int kc = 0; kc < kM / 32; ++kc) {
        v16bf a = load_A_tile(Xe, nt * 16, kc * 32, kM, lane);
        v16bf b = load_B_tile(pW1, (2 * kP) / 32 + kc, ht, kM / 16, lane);  // X-part of W1
        acc = wmma_bf16(a, b, acc);
    }
    int col = ht * 16 + (lane & 15);
    float bias = b1[col];
    int r0 = nt * 16 + ((lane & 16) ? 8 : 0);
    #pragma unroll
    for (int v = 0; v < 8; ++v)
        U[(size_t)(r0 + v) * kM + col] = f2bf(acc[v] + bias);
}

// ---------- per-step kernels ----------
// S[b][h] = sum_{k<512} [d|c][b][k] * W1[h][k]
__global__ __launch_bounds__(256) void k_gemm_S(const bf16_t* __restrict__ dc,
                                                const u32* __restrict__ pW1,
                                                float* __restrict__ S) {
    int gw = (blockIdx.x * 256 + threadIdx.x) >> 5;
    int lane = threadIdx.x & 31;
    const int numHT = kM / 16;
    int bt = gw / numHT;
    int ht = gw - bt * numHT;
    v8f acc = {};
    #pragma unroll
    for (int kc = 0; kc < (2 * kP) / 32; ++kc) {
        v16bf a = load_A_tile(dc, bt * 16, kc * 32, 2 * kP, lane);
        v16bf b = load_B_tile(pW1, kc, ht, kM / 16, lane);
        acc = wmma_bf16(a, b, acc);
    }
    int col = ht * 16 + (lane & 15);
    int r0 = bt * 16 + ((lane & 16) ? 8 : 0);
    #pragma unroll
    for (int v = 0; v < 8; ++v)
        S[(size_t)(r0 + v) * kM + col] = acc[v];
}

// Fused: scores = W2.tanh(U[b,t,:]+S[b,:]) -> softmax -> context -> y_tilde
__global__ __launch_bounds__(256) void k_attn_step(
        const bf16_t* __restrict__ U, const bf16_t* __restrict__ Xe,
        const float* __restrict__ S,  const float* __restrict__ W2,
        const float* __restrict__ fcW, const float* __restrict__ fcb,
        const float* __restrict__ yprev, int t,
        float* __restrict__ ctx, float* __restrict__ yt) {
    __shared__ float sc[128];
    __shared__ float red[8];
    int b = blockIdx.x;
    int tid = threadIdx.x;
    int wave = tid >> 5, lane = tid & 31;
    int h0 = lane * 8;
    float sreg[8], wreg[8];
    const float* Sb = S + (size_t)b * kM;
    #pragma unroll
    for (int i = 0; i < 8; ++i) { sreg[i] = Sb[h0 + i]; wreg[i] = W2[h0 + i]; }

    const bf16_t* Ub = U + (size_t)b * kT * kM;
    for (int tt = wave; tt < kT; tt += 8) {
        union { uint4 u; bf16_t h[8]; } cv;
        cv.u = *(const uint4*)(Ub + (size_t)tt * kM + h0);
        float a = 0.f;
        #pragma unroll
        for (int i = 0; i < 8; ++i) a += wreg[i] * tanhf(bf2f(cv.h[i]) + sreg[i]);
        #pragma unroll
        for (int o = 16; o; o >>= 1) a += __shfl_xor(a, o, 32);
        if (lane == 0) sc[tt] = a;
    }
    __syncthreads();
    // softmax over kT scores
    float v = (tid < kT) ? sc[tid] : -3.4e38f;
    float m = v;
    #pragma unroll
    for (int o = 16; o; o >>= 1) m = fmaxf(m, __shfl_xor(m, o, 32));
    if (lane == 0) red[wave] = m;
    __syncthreads();
    float mall = red[0];
    #pragma unroll
    for (int w = 1; w < 8; ++w) mall = fmaxf(mall, red[w]);
    float e = (tid < kT) ? __expf(v - mall) : 0.f;
    float ss = e;
    #pragma unroll
    for (int o = 16; o; o >>= 1) ss += __shfl_xor(ss, o, 32);
    __syncthreads();
    if (lane == 0) red[wave] = ss;
    __syncthreads();
    float tot = 0.f;
    #pragma unroll
    for (int w = 0; w < 8; ++w) tot += red[w];
    if (tid < kT) sc[tid] = e / tot;
    __syncthreads();
    // context[b][tid] = sum_t beta[t] * X[b,t,tid]   (bf16 X copy, L2-resident)
    const bf16_t* Xb = Xe + (size_t)b * kT * kM;
    float cacc = 0.f;
    for (int tt = 0; tt < kT; ++tt)
        cacc = fmaf(sc[tt], bf2f(Xb[(size_t)tt * kM + tid]), cacc);
    ctx[(size_t)b * kM + tid] = cacc;
    // y_tilde[b] = fc_W[0:M].context + fc_W[M]*y_prev[b,t] + fc_b
    float part = cacc * fcW[tid];
    #pragma unroll
    for (int o = 16; o; o >>= 1) part += __shfl_xor(part, o, 32);
    if (lane == 0) red[wave] = part;
    __syncthreads();
    if (tid == 0) {
        float s2 = 0.f;
        #pragma unroll
        for (int w = 0; w < 8; ++w) s2 += red[w];
        yt[b] = s2 + fcW[kM] * yprev[(size_t)b * kT + t] + fcb[0];
    }
}

// Fused LSTM: gates = d@W_hh^T (WMMA) + y_tilde*W_ih + b, then cell update.
// One block per 16 batch rows; gates kept in 64KB LDS (WGP has 320KB).
__global__ __launch_bounds__(256) void k_lstm_step(
        const bf16_t* __restrict__ dcin, const u32* __restrict__ pWhh,
        const float* __restrict__ Wih, const float* __restrict__ bih,
        const float* __restrict__ bhh, const float* __restrict__ yt,
        float* __restrict__ cst, float* __restrict__ dst, bf16_t* __restrict__ dcout) {
    __shared__ float gates[16 * 4 * kP];   // 16 rows x 1024 gate cols
    int b0 = blockIdx.x * 16;
    int tid = threadIdx.x, wave = tid >> 5, lane = tid & 31;
    for (int i = 0; i < 8; ++i) {          // 8 n-tiles per wave -> 64 tiles = 1024 cols
        int nt = wave * 8 + i;
        v8f acc = {};
        #pragma unroll
        for (int kc = 0; kc < kP / 32; ++kc) {
            v16bf a  = load_A_tile(dcin, b0, kc * 32, 2 * kP, lane);   // d part: k<256
            v16bf bm = load_B_tile(pWhh, kc, nt, (4 * kP) / 16, lane);
            acc = wmma_bf16(a, bm, acc);
        }
        int col = nt * 16 + (lane & 15);
        int r0 = (lane & 16) ? 8 : 0;
        #pragma unroll
        for (int v = 0; v < 8; ++v) gates[(r0 + v) * (4 * kP) + col] = acc[v];
    }
    __syncthreads();
    for (int e = tid; e < 16 * kP; e += 256) {
        int r = e >> 8;
        int p = e & (kP - 1);
        int b = b0 + r;
        float y = yt[b];
        const float* g = gates + r * (4 * kP);
        float gi = g[p]           + y * Wih[p]           + bih[p]           + bhh[p];
        float gf = g[kP + p]      + y * Wih[kP + p]      + bih[kP + p]      + bhh[kP + p];
        float gg = g[2 * kP + p]  + y * Wih[2 * kP + p]  + bih[2 * kP + p]  + bhh[2 * kP + p];
        float go = g[3 * kP + p]  + y * Wih[3 * kP + p]  + bih[3 * kP + p]  + bhh[3 * kP + p];
        float cold = cst[(size_t)b * kP + p];
        float cn = sigf(gf) * cold + sigf(gi) * tanhf(gg);
        float dn = sigf(go) * tanhf(cn);
        cst[(size_t)b * kP + p] = cn;
        dst[(size_t)b * kP + p] = dn;
        dcout[(size_t)b * 2 * kP + p]      = f2bf(dn);   // next-step WMMA A operand
        dcout[(size_t)b * 2 * kP + kP + p] = f2bf(cn);
    }
}

__global__ __launch_bounds__(256) void k_final(const float* __restrict__ dst,
                                               const float* __restrict__ ctx,
                                               const float* __restrict__ fcfW,
                                               const float* __restrict__ fcfb,
                                               float* __restrict__ out) {
    __shared__ float red[8];
    int b = blockIdx.x, tid = threadIdx.x, wave = tid >> 5, lane = tid & 31;
    float part = dst[(size_t)b * kP + tid] * fcfW[tid]
               + ctx[(size_t)b * kM + tid] * fcfW[kP + tid];
    #pragma unroll
    for (int o = 16; o; o >>= 1) part += __shfl_xor(part, o, 32);
    if (lane == 0) red[wave] = part;
    __syncthreads();
    if (tid == 0) {
        float s = 0.f;
        #pragma unroll
        for (int w = 0; w < 8; ++w) s += red[w];
        out[b] = s + fcfb[0];
    }
}

extern "C" void kernel_launch(void* const* d_in, const int* in_sizes, int n_in,
                              void* d_out, int out_size, void* d_ws, size_t ws_size,
                              hipStream_t stream) {
    (void)in_sizes; (void)n_in; (void)out_size; (void)ws_size;
    const float* X    = (const float*)d_in[0];
    const float* yprv = (const float*)d_in[1];
    const float* W1   = (const float*)d_in[2];
    const float* b1   = (const float*)d_in[3];
    const float* W2   = (const float*)d_in[4];
    /* d_in[5] = attn_b2: softmax-invariant constant, omitted */
    const float* Wih  = (const float*)d_in[6];
    const float* Whh  = (const float*)d_in[7];
    const float* bih  = (const float*)d_in[8];
    const float* bhh  = (const float*)d_in[9];
    const float* fcW  = (const float*)d_in[10];
    const float* fcb  = (const float*)d_in[11];
    const float* fcfW = (const float*)d_in[12];
    const float* fcfb = (const float*)d_in[13];
    float* out = (float*)d_out;

    char* ws = (char*)d_ws;
    size_t off = 0;
    auto take = [&](size_t bytes) {
        char* p = ws + off;
        off += (bytes + 255) & ~(size_t)255;
        return p;
    };
    bf16_t* Xe   = (bf16_t*)take((size_t)kNT * kM * 2);        // 66.6 MB (L2-resident)
    bf16_t* U    = (bf16_t*)take((size_t)kNT * kM * 2);        // 66.6 MB (L2-resident)
    u32*    pW1  = (u32*)   take((size_t)kKW1 * kM * 2);       // packed W1 (bf16)
    u32*    pWhh = (u32*)   take((size_t)kP * 4 * kP * 2);     // packed W_hh (bf16)
    bf16_t* dc   = (bf16_t*)take((size_t)kB * 2 * kP * 2);     // [d|c] bf16 state
    float*  dst_ = (float*) take((size_t)kB * kP * 4);         // d state f32
    float*  cst  = (float*) take((size_t)kB * kP * 4);         // c state f32
    float*  S    = (float*) take((size_t)kB * kM * 4);         // per-step d/c attn bias
    float*  ctx  = (float*) take((size_t)kB * kM * 4);         // context
    float*  yt   = (float*) take((size_t)kB * 4);              // y_tilde

    // zero-init recurrent state (graph-capture-safe)
    hipMemsetAsync(dc,   0, (size_t)kB * 2 * kP * 2, stream);
    hipMemsetAsync(dst_, 0, (size_t)kB * kP * 4, stream);
    hipMemsetAsync(cst,  0, (size_t)kB * kP * 4, stream);

    // one-time: bf16 X copy, weight packing, step-invariant U = X@W1c^T + b1
    { int n = kNT * kM;
      k_convert_bf16<<<(n + 255) / 256, 256, 0, stream>>>(X, Xe, n); }
    k_pack_B<<<(((kKW1 >> 1) * kM) + 255) / 256, 256, 0, stream>>>(W1, pW1, kKW1, kM);
    k_pack_B<<<(((kP >> 1) * 4 * kP) + 255) / 256, 256, 0, stream>>>(Whh, pWhh, kP, 4 * kP);
    { int waves = (kNT / 16) * (kM / 16);
      k_gemm_U<<<waves / 8, 256, 0, stream>>>(Xe, pW1, b1, U); }

    // 127 sequential decoder steps
    for (int t = 0; t < kT; ++t) {
        { int waves = (kB / 16) * (kM / 16);
          k_gemm_S<<<waves / 8, 256, 0, stream>>>(dc, pW1, S); }
        k_attn_step<<<kB, 256, 0, stream>>>(U, Xe, S, W2, fcW, fcb, yprv, t, ctx, yt);
        k_lstm_step<<<kB / 16, 256, 0, stream>>>(dc, pWhh, Wih, bih, bhh, yt, cst, dst_, dc);
    }
    k_final<<<kB, 256, 0, stream>>>(dst_, ctx, fcfW, fcfb, out);
}
